// Node2Vec_65859028517221
// MI455X (gfx1250) — matrix operand order repaired and checked by reading, not compiled, not run on hardware
//
#include <hip/hip_runtime.h>

typedef __attribute__((ext_vector_type(16))) _Float16 v16h;
typedef __attribute__((ext_vector_type(8)))  float    v8f;

#define NNODES   100000
#define DIM      64
#define LWALK    13
#define KWIN     10
#define NSAMP    5
#define WPR      4                    // 1 + L - K
#define NWIN     (NNODES * WPR)       // 400000 windows
#define NTILES   (NWIN / 16)          // 25000 tiles of 16 windows
#define NSLOT    (KWIN - 1 + NSAMP)   // 9 pos + 5 neg = 14
#define EPSV     1e-8f
#define LN2F     0.69314718056f

#define WAVES_PER_BLOCK 8
#define THREADS (WAVES_PER_BLOCK * 32)
#define NBLOCKS 1024

__global__ void n2v_init(float* ws) {
  ws[0] = 0.0f;   // sum of positive losses
  ws[1] = 0.0f;   // sum of negative losses
}

// 16 contiguous floats -> 16 halves of a v16h (4x b128 load + 8x v_cvt_pk).
__device__ __forceinline__ void cvt16(const float* __restrict__ p, v16h& dst) {
  const float4 f0 = *reinterpret_cast<const float4*>(p + 0);
  const float4 f1 = *reinterpret_cast<const float4*>(p + 4);
  const float4 f2 = *reinterpret_cast<const float4*>(p + 8);
  const float4 f3 = *reinterpret_cast<const float4*>(p + 12);
  dst[0]=(_Float16)f0.x; dst[1]=(_Float16)f0.y; dst[2]=(_Float16)f0.z; dst[3]=(_Float16)f0.w;
  dst[4]=(_Float16)f1.x; dst[5]=(_Float16)f1.y; dst[6]=(_Float16)f1.z; dst[7]=(_Float16)f1.w;
  dst[8]=(_Float16)f2.x; dst[9]=(_Float16)f2.y; dst[10]=(_Float16)f2.z; dst[11]=(_Float16)f2.w;
  dst[12]=(_Float16)f3.x; dst[13]=(_Float16)f3.y; dst[14]=(_Float16)f3.z; dst[15]=(_Float16)f3.w;
}

__global__ __launch_bounds__(THREADS)
void n2v_main(const int* __restrict__ rw, const int* __restrict__ neg,
              const float* __restrict__ emb, float* __restrict__ ws) {
  const int lane = threadIdx.x & 31;
  const int wave = threadIdx.x >> 5;

  const int waveId = blockIdx.x * WAVES_PER_BLOCK + wave;
  const int nWaves = gridDim.x * WAVES_PER_BLOCK;

  const int j16 = lane & 15;            // window/row/column owned by this lane
  const int hi  = lane >> 4;            // lane-group (K-split select)
  const int kbA = hi * 8;               // unused; A fragments load at kbB, like B
  const int kbB = hi * 16;              // B-fragment K base within 32-chunk

  const int sel = (lane < 8) ? lane : (lane - 24);             // diag VGPR index
  // fold diag-holder lane mask AND the (-ln2) base-2 -> natural-log conversion
  // into one per-lane constant: loss contribution = mneg * log2(arg)
  const float mneg = ((lane < 8) || (lane >= 24)) ? -LN2F : 0.0f;

  float accPos = 0.0f, accNeg = 0.0f;

  for (int tile = waveId; tile < NTILES; tile += nWaves) {
    const int win  = tile * 16 + j16;   // this lane's window (dup across hi groups)
    const int node = win >> 2;
    const int off  = win & 3;
    const int si   = rw[node * LWALK + off];   // start node index

    // ---- A fragments gathered directly from global (L2-resident) ----
    // Lane reads emb[si] runs {kbB..kbB+15} (chunk0), +32 (chunk1).
    // Lanes l / l+16 cover disjoint halves -> each row read exactly once.
    const float* xrow = emb + si * DIM;
    v16h a0, a1;
    {
      const float4 p0 = *reinterpret_cast<const float4*>(xrow + kbB);
      const float4 p1 = *reinterpret_cast<const float4*>(xrow + kbB + 4);
      const float4 p2 = *reinterpret_cast<const float4*>(xrow + kbB + 8);
      const float4 p3 = *reinterpret_cast<const float4*>(xrow + kbB + 12);
      const float4 p4 = *reinterpret_cast<const float4*>(xrow + 32 + kbB);
      const float4 p5 = *reinterpret_cast<const float4*>(xrow + 32 + kbB + 4);
      const float4 p6 = *reinterpret_cast<const float4*>(xrow + 32 + kbB + 8);
      const float4 p7 = *reinterpret_cast<const float4*>(xrow + 32 + kbB + 12);
      a0[0]=(_Float16)p0.x; a0[1]=(_Float16)p0.y; a0[2]=(_Float16)p0.z; a0[3]=(_Float16)p0.w;
      a0[4]=(_Float16)p1.x; a0[5]=(_Float16)p1.y; a0[6]=(_Float16)p1.z; a0[7]=(_Float16)p1.w;
      a0[8]=(_Float16)p2.x; a0[9]=(_Float16)p2.y; a0[10]=(_Float16)p2.z; a0[11]=(_Float16)p2.w;
      a0[12]=(_Float16)p3.x; a0[13]=(_Float16)p3.y; a0[14]=(_Float16)p3.z; a0[15]=(_Float16)p3.w;
      a1[0]=(_Float16)p4.x; a1[1]=(_Float16)p4.y; a1[2]=(_Float16)p4.z; a1[3]=(_Float16)p4.w;
      a1[4]=(_Float16)p5.x; a1[5]=(_Float16)p5.y; a1[6]=(_Float16)p5.z; a1[7]=(_Float16)p5.w;
      a1[8]=(_Float16)p6.x; a1[9]=(_Float16)p6.y; a1[10]=(_Float16)p6.z; a1[11]=(_Float16)p6.w;
      a1[12]=(_Float16)p7.x; a1[13]=(_Float16)p7.y; a1[14]=(_Float16)p7.z; a1[15]=(_Float16)p7.w;
    }

    #pragma unroll 1
    for (int s = 0; s < NSLOT; ++s) {
      int ci;
      if (s < KWIN - 1) ci = rw[node * LWALK + off + 1 + s];      // positive context
      else              ci = neg[win * NSAMP + (s - (KWIN - 1))];  // negative sample

      // ---- B fragments: column j16 = context of window j16 ----
      // Lane reads emb[ci] runs {kbB..kbB+15} (chunk0) and +32 (chunk1):
      // contiguous 64B per lane; row covered once by lanes l / l+16.
      const float* crow = emb + ci * DIM;
      v16h b0, b1;
      cvt16(crow + kbB,      b0);
      cvt16(crow + 32 + kbB, b1);

      // D = X_start(16x64) * Ctx^T(64x16), f32 accumulate over two K=32 chunks
      v8f c = {};
      c = __builtin_amdgcn_wmma_f32_16x16x32_f16(false, a0, false, b0,
                                                 (short)0, c, false, false);
      c = __builtin_amdgcn_wmma_f32_16x16x32_f16(false, a1, false, b1,
                                                 (short)0, c, false, false);

      // diagonal extraction: lanes 0-7 -> c[lane]; lanes 24-31 -> c[lane-24]
      float dv = c[0];
      #pragma unroll
      for (int e = 1; e < 8; ++e) if (sel == e) dv = c[e];

      // sigmoid via single v_rcp_f32; raw v_log_f32 (args in [1e-8, ~1], always
      // normal-range, so the libm denorm guard is unnecessary); mask+(-ln2)
      // pre-folded into mneg -> post-sigmoid cost = 1 trans + 1 fma.
      const float sg = __builtin_amdgcn_rcpf(1.0f + __expf(-dv));
      if (s < KWIN - 1) {                  // wave-uniform branch (scalar)
        accPos = fmaf(mneg, __builtin_amdgcn_logf(sg + EPSV), accPos);
      } else {
        accNeg = fmaf(mneg, __builtin_amdgcn_logf(1.0f - sg + EPSV), accNeg);
      }
    }
  }

  __hip_atomic_fetch_add(&ws[0], accPos, __ATOMIC_RELAXED, __HIP_MEMORY_SCOPE_AGENT);
  __hip_atomic_fetch_add(&ws[1], accNeg, __ATOMIC_RELAXED, __HIP_MEMORY_SCOPE_AGENT);
}

__global__ void n2v_final(const float* __restrict__ ws, float* __restrict__ out) {
  out[0] = ws[0] / (float)(NWIN * (KWIN - 1)) + ws[1] / (float)(NWIN * NSAMP);
}

extern "C" void kernel_launch(void* const* d_in, const int* in_sizes, int n_in,
                              void* d_out, int out_size, void* d_ws, size_t ws_size,
                              hipStream_t stream) {
  (void)in_sizes; (void)n_in; (void)out_size; (void)ws_size;
  const int*   rw  = (const int*)d_in[0];
  const int*   neg = (const int*)d_in[1];
  const float* emb = (const float*)d_in[2];
  float*       out = (float*)d_out;
  float*       ws  = (float*)d_ws;

  n2v_init<<<1, 1, 0, stream>>>(ws);
  n2v_main<<<NBLOCKS, THREADS, 0, stream>>>(rw, neg, emb, ws);
  n2v_final<<<1, 1, 0, stream>>>(ws, out);
}
